// build_model_25048249270471
// MI455X (gfx1250) — compile-verified, matrix operand order checked by reading, not run on hardware
//
#include <hip/hip_runtime.h>
#include <cstdint>

// ---------------- model constants ----------------
#define Bsz   4
#define Sq    1024
#define Dm    1024
#define Hh    16
#define HDim  64
#define Ll    6
#define OUTd  512
#define BS    (Bsz * Sq)     // 4096 rows in the residual stream

typedef __bf16 bf16;
typedef __attribute__((ext_vector_type(16))) __bf16 v16bf;
typedef __attribute__((ext_vector_type(8)))  __bf16 v8bf;
typedef __attribute__((ext_vector_type(8)))  float  v8f;
typedef __attribute__((ext_vector_type(4)))  unsigned int v4u;
typedef __attribute__((ext_vector_type(8)))  int v8i;
typedef __attribute__((ext_vector_type(4)))  int v4i;

// ---------------- WMMA helpers ----------------
__device__ __forceinline__ v8f wmma_bf16(v16bf a, v16bf b, v8f c) {
  // D = A(16x32) * B(32x16) + C, f32 accumulate
  return __builtin_amdgcn_wmma_f32_16x16x32_bf16(
      /*neg_a=*/false, a, /*neg_b=*/false, b,
      /*c_mod=*/(short)0, c, /*reuse_a=*/false, /*reuse_b=*/false);
}

// Per-lane 16-bit fragment load (A layout; B uses the same pattern on Wt rows /
// K rows / V^T rows). Lane l: row = l&15, k-base = (l>=16)?8:0; VGPR0..3 hold
// k-pairs {kb+0..7}, VGPR4..7 hold {kb+16..23}  (CDNA5 ISA 7.12.2).
__device__ __forceinline__ v16bf load_frag(const bf16* __restrict__ base, int ld) {
  int lane = threadIdx.x & 31;
  const bf16* p = base + (size_t)(lane & 15) * (size_t)ld + ((lane >> 4) << 3);
  v8bf lo = *reinterpret_cast<const v8bf*>(p);        // k-run [kb .. kb+7]
  v8bf hi = *reinterpret_cast<const v8bf*>(p + 16);   // k-run [kb+16 .. kb+23]
  return __builtin_shufflevector(lo, hi, 0,1,2,3,4,5,6,7,8,9,10,11,12,13,14,15);
}

__device__ __forceinline__ unsigned rfl(unsigned v) {
  return (unsigned)__builtin_amdgcn_readfirstlane((int)v);
}

// ---------------- elementwise kernels ----------------
__global__ __launch_bounds__(256) void cvt_f32_bf16(const float* __restrict__ in,
                                                    bf16* __restrict__ out, int n) {
  int i = blockIdx.x * 256 + threadIdx.x;
  if (i < n) out[i] = (bf16)in[i];
}

__global__ __launch_bounds__(256) void add_pe(const float* __restrict__ x,
                                              const float* __restrict__ pe,
                                              float* __restrict__ h, int n) {
  int i = blockIdx.x * 256 + threadIdx.x;
  if (i < n) h[i] = x[i] + pe[i % (Sq * Dm)];
}

// ---------------- layernorm (f32 in -> bf16 out) ----------------
__global__ __launch_bounds__(256) void layernorm_bf16(const float* __restrict__ X,
                                                      const float* __restrict__ g,
                                                      const float* __restrict__ bta,
                                                      bf16* __restrict__ Y) {
  int row = blockIdx.x;
  const float* x = X + (size_t)row * Dm;
  float s = 0.f, s2 = 0.f;
  for (int i = threadIdx.x; i < Dm; i += 256) { float v = x[i]; s += v; s2 += v * v; }
  for (int m = 1; m < 32; m <<= 1) { s += __shfl_xor(s, m, 32); s2 += __shfl_xor(s2, m, 32); }
  __shared__ float sbuf[16];
  int wid = threadIdx.x >> 5;
  if ((threadIdx.x & 31) == 0) { sbuf[wid] = s; sbuf[8 + wid] = s2; }
  __syncthreads();
  float ts = 0.f, ts2 = 0.f;
#pragma unroll
  for (int w = 0; w < 8; ++w) { ts += sbuf[w]; ts2 += sbuf[8 + w]; }
  float mu  = ts  * (1.0f / Dm);
  float var = ts2 * (1.0f / Dm) - mu * mu;
  float r   = rsqrtf(var + 1e-5f);
  for (int i = threadIdx.x; i < Dm; i += 256)
    Y[(size_t)row * Dm + i] = (bf16)((x[i] - mu) * r * g[i] + bta[i]);
}

// ---------------- WMMA GEMM: Out = A[MxK] * Wt[NxK]^T (+ residual) --------
// 8 waves / block; wave -> 16 rows x 64 cols (4 accumulators), K-step = 32.
// The 64x32 bf16 weight tile shared by all 8 waves is staged into LDS by the
// Tensor Data Mover (wave 0 issues tensor_load_to_lds, TENSORcnt-tracked),
// double-buffered so the next tile's DMA overlaps this tile's WMMAs.
// mode 0: f32 out = res + acc   mode 1: bf16 out
// mode 2: bf16 out transposed for V -> [B,H,HD,S]   mode 3: f32 out
__global__ __launch_bounds__(256) void gemm_wmma(const bf16* __restrict__ A,
                                                 const bf16* __restrict__ Wt,
                                                 const float* __restrict__ Res,
                                                 void* __restrict__ Out,
                                                 int M, int N, int K, int mode) {
  __shared__ __align__(16) bf16 bstage[2][64 * 32];   // 2 x 4KB weight tiles
  int wv   = threadIdx.x >> 5;
  int lane = threadIdx.x & 31;
  int m0 = blockIdx.y * 128 + wv * 16;
  int n0 = blockIdx.x * 64;

  // LDS byte offset of the staging buffer (addrspace(3) ptr == LDS offset)
  unsigned lds_base = (unsigned)(size_t)(__attribute__((address_space(3))) char*)
                          (&bstage[0][0]);

  // Invariant descriptor group 1: data_size=2B, tensor_dim0=K, tensor_dim1=64,
  // tile = 32(x) x 64(y), dim0 stride = K elements.
  int g1d0 = (int)rfl(0x00010000u);                       // data_size=1 (2 bytes)
  int g1d1 = (int)rfl((unsigned)((K & 0xFFFF) << 16));    // tensor_dim0[15:0]
  int g1d2 = (int)rfl((unsigned)(((unsigned)K >> 16) & 0xFFFFu) | (64u << 16)); // dim0 hi | dim1 lo
  int g1d3 = (int)rfl(32u << 16);                         // dim1 hi=0 | tile_dim0=32
  int g1d4 = (int)rfl(64u);                               // tile_dim1=64, tile_dim2=0
  int g1d5 = (int)rfl((unsigned)K);                       // tensor_dim0_stride lo32
  v8i g1 = { g1d0, g1d1, g1d2, g1d3, g1d4, g1d5, 0, 0 };
  v4i gz = { 0, 0, 0, 0 };
  v8i gz8 = { 0, 0, 0, 0, 0, 0, 0, 0 };

  v8f acc[4] = {v8f{}, v8f{}, v8f{}, v8f{}};
  const bf16* Abase = A + (size_t)m0 * K;
  const bf16* Wbase = Wt + (size_t)n0 * K;

  // Prologue: DMA tile k=0 into buffer 0
  if (threadIdx.x < 32) {
    unsigned long long ga = (unsigned long long)(uintptr_t)Wbase;
    v4u g0 = { rfl(1u),                                   // count=1 valid descriptor
               rfl(lds_base),
               rfl((unsigned)(ga & 0xFFFFFFFFu)),
               rfl((unsigned)((ga >> 32) & 0x01FFFFFFu) | 0x80000000u) }; // type=2
    __builtin_amdgcn_tensor_load_to_lds(g0, g1, gz, gz, gz8, 0);
  }

  int p = 0;
  for (int k = 0; k < K; k += 32) {
    // Prefetch next tile into the other buffer before waiting on this one.
    if (threadIdx.x < 32) {
      if (k + 32 < K) {
        unsigned long long ga = (unsigned long long)(uintptr_t)(Wbase + k + 32);
        v4u g0 = { rfl(1u),
                   rfl(lds_base + (unsigned)((p ^ 1) * (64 * 32 * 2))),
                   rfl((unsigned)(ga & 0xFFFFFFFFu)),
                   rfl((unsigned)((ga >> 32) & 0x01FFFFFFu) | 0x80000000u) };
        __builtin_amdgcn_tensor_load_to_lds(g0, g1, gz, gz, gz8, 0);
        __builtin_amdgcn_s_wait_tensorcnt(1);   // current tile done, next in flight
      } else {
        __builtin_amdgcn_s_wait_tensorcnt(0);
      }
    }
    __syncthreads();   // tile p ready for all waves; also fences buffer reuse

    v16bf af = load_frag(Abase + k, K);
#pragma unroll
    for (int t = 0; t < 4; ++t) {
      v16bf bfrag = load_frag(&bstage[p][t * 16 * 32], 32);  // ds_load_b128 pair
      acc[t] = wmma_bf16(af, bfrag, acc[t]);
    }
    p ^= 1;
  }

  int nl = lane & 15, hl = lane >> 4;
#pragma unroll
  for (int t = 0; t < 4; ++t)
#pragma unroll
    for (int r = 0; r < 8; ++r) {
      int row = m0 + hl * 8 + r;          // C layout: VGPR r -> M=r (+8 for hi half)
      int col = n0 + t * 16 + nl;
      float v = acc[t][r];
      if (mode == 0) {
        ((float*)Out)[(size_t)row * N + col] = Res[(size_t)row * N + col] + v;
      } else if (mode == 1) {
        ((bf16*)Out)[(size_t)row * N + col] = (bf16)v;
      } else if (mode == 2) {             // V^T: [B,H,HD,S]
        int bb = row >> 10, ss = row & 1023;
        int hh = col >> 6,  hd = col & 63;
        ((bf16*)Out)[(((size_t)(bb * Hh + hh)) * HDim + hd) * Sq + ss] = (bf16)v;
      } else {
        ((float*)Out)[(size_t)row * N + col] = v;
      }
    }
}

// ---------------- flash attention: 1 wave per (b,h,16-query tile) ----------
__global__ __launch_bounds__(32) void attn_wmma(const bf16* __restrict__ Q,
                                                const bf16* __restrict__ Kd,
                                                const bf16* __restrict__ VT,
                                                bf16* __restrict__ Oc) {
  __shared__ __align__(16) bf16 pbuf[16 * 32];   // P tile (16 q rows x 32 keys)
  int lane = threadIdx.x;
  int nl = lane & 15, hl = lane >> 4;
  int q0 = blockIdx.x * 16;
  int bh = blockIdx.y;
  int b = bh / Hh, hh = bh % Hh;
  const float scale = 0.125f;                    // 1/sqrt(64)

  const bf16* qb = Q + ((size_t)(b * Sq + q0)) * Dm + hh * HDim;
  v16bf qf0 = load_frag(qb,      Dm);            // hd 0..31
  v16bf qf1 = load_frag(qb + 32, Dm);            // hd 32..63

  v8f o[4] = {v8f{}, v8f{}, v8f{}, v8f{}};
  float rowmax[8], rowsum[8];
#pragma unroll
  for (int r = 0; r < 8; ++r) { rowmax[r] = -INFINITY; rowsum[r] = 0.f; }

  int nkt = (q0 + 16 + 31) / 32;                 // causal: keys 0 .. q0+15
  for (int jt = 0; jt < nkt; ++jt) {
    int k0 = jt * 32;
    const bf16* kb = Kd + ((size_t)(b * Sq + k0)) * Dm + hh * HDim;

    // scores: two 16-key subtiles, K-dim = HD = 64 (2 WMMA steps each)
    v8f s0 = v8f{}, s1 = v8f{};
    s0 = wmma_bf16(qf0, load_frag(kb,              Dm), s0);
    s0 = wmma_bf16(qf1, load_frag(kb + 32,         Dm), s0);
    s1 = wmma_bf16(qf0, load_frag(kb + 16 * Dm,      Dm), s1);
    s1 = wmma_bf16(qf1, load_frag(kb + 16 * Dm + 32, Dm), s1);

    float scl[8];
#pragma unroll
    for (int r = 0; r < 8; ++r) {
      int qrow = q0 + hl * 8 + r;
      float e0 = s0[r] * scale;  if (k0 + nl      > qrow) e0 = -INFINITY;
      float e1 = s1[r] * scale;  if (k0 + 16 + nl > qrow) e1 = -INFINITY;
      float mx = fmaxf(e0, e1);
#pragma unroll
      for (int m = 1; m < 16; m <<= 1) mx = fmaxf(mx, __shfl_xor(mx, m, 32));
      float mnew = fmaxf(rowmax[r], mx);
      float sc = __expf(rowmax[r] - mnew);
      rowmax[r] = mnew;
      float p0 = __expf(e0 - mnew);
      float p1 = __expf(e1 - mnew);
      float ps = p0 + p1;
#pragma unroll
      for (int m = 1; m < 16; m <<= 1) ps += __shfl_xor(ps, m, 32);
      rowsum[r] = rowsum[r] * sc + ps;
      scl[r] = sc;
      pbuf[(hl * 8 + r) * 32 + nl]      = (bf16)p0;
      pbuf[(hl * 8 + r) * 32 + 16 + nl] = (bf16)p1;
    }
#pragma unroll
    for (int t = 0; t < 4; ++t)
#pragma unroll
      for (int r = 0; r < 8; ++r) o[t][r] *= scl[r];

    // CDNA5 split-counter wait: LDS stores visible before cross-lane reload
    asm volatile("s_wait_dscnt 0x0" ::: "memory");

    // reload P as an A fragment from LDS (ds_load_b128 pair)
    const bf16* pp = &pbuf[(size_t)nl * 32 + (hl << 3)];
    v8bf plo = *reinterpret_cast<const v8bf*>(pp);
    v8bf phi = *reinterpret_cast<const v8bf*>(pp + 16);
    v16bf pf = __builtin_shufflevector(plo, phi,
                 0,1,2,3,4,5,6,7,8,9,10,11,12,13,14,15);

    // P(16x32 keys) * V(32 keys x 64 hd); V^T rows are hd, contiguous keys
    const bf16* vb = VT + ((size_t)(b * Hh + hh)) * HDim * Sq + k0;
#pragma unroll
    for (int t = 0; t < 4; ++t)
      o[t] = wmma_bf16(pf, load_frag(vb + (size_t)(t * 16) * Sq, Sq), o[t]);

    asm volatile("s_wait_dscnt 0x0" ::: "memory");
  }

#pragma unroll
  for (int t = 0; t < 4; ++t)
#pragma unroll
    for (int r = 0; r < 8; ++r) {
      int qrow = q0 + hl * 8 + r;
      float v = o[t][r] / rowsum[r];
      Oc[((size_t)(b * Sq + qrow)) * Dm + hh * HDim + t * 16 + nl] = (bf16)v;
    }
}

// ---------------- launcher ----------------
extern "C" void kernel_launch(void* const* d_in, const int* in_sizes, int n_in,
                              void* d_out, int out_size, void* d_ws, size_t ws_size,
                              hipStream_t stream) {
  const float* x     = (const float*)d_in[0];
  const float* pe    = (const float*)d_in[1];
  // d_in[2] = causal mask, applied analytically in attn_wmma
  const float* ln1_g = (const float*)d_in[3];
  const float* ln1_b = (const float*)d_in[4];
  const float* wq    = (const float*)d_in[5];
  const float* wk    = (const float*)d_in[6];
  const float* wv    = (const float*)d_in[7];
  const float* wo    = (const float*)d_in[8];
  const float* ln2_g = (const float*)d_in[9];
  const float* ln2_b = (const float*)d_in[10];
  const float* wf    = (const float*)d_in[11];
  const float* lnf_g = (const float*)d_in[12];
  const float* lnf_b = (const float*)d_in[13];
  const float* w_out = (const float*)d_in[14];
  float* out = (float*)d_out;

  char* wsb = (char*)d_ws;
  size_t off = 0;
  auto alloc = [&](size_t bytes) -> void* {
    void* p = wsb + off;
    off += (bytes + 255) & ~(size_t)255;
    return p;
  };
  const size_t WSZ = (size_t)Ll * Dm * Dm;       // elements per weight stack
  bf16* wq_b   = (bf16*)alloc(WSZ * 2);
  bf16* wk_b   = (bf16*)alloc(WSZ * 2);
  bf16* wv_b   = (bf16*)alloc(WSZ * 2);
  bf16* wo_b   = (bf16*)alloc(WSZ * 2);
  bf16* wf_b   = (bf16*)alloc(WSZ * 2);
  bf16* wout_b = (bf16*)alloc((size_t)OUTd * Dm * 2);
  float* h     = (float*)alloc((size_t)BS * Dm * 4);
  bf16* hn     = (bf16*)alloc((size_t)BS * Dm * 2);
  bf16* qb     = (bf16*)alloc((size_t)BS * Dm * 2);
  bf16* kb     = (bf16*)alloc((size_t)BS * Dm * 2);
  bf16* vT     = (bf16*)alloc((size_t)BS * Dm * 2);
  bf16* attnb  = (bf16*)alloc((size_t)BS * Dm * 2);

  // weights -> bf16 (once per launch; deterministic)
  int nw = (int)WSZ;
  dim3 cg((nw + 255) / 256);
  cvt_f32_bf16<<<cg, 256, 0, stream>>>(wq, wq_b, nw);
  cvt_f32_bf16<<<cg, 256, 0, stream>>>(wk, wk_b, nw);
  cvt_f32_bf16<<<cg, 256, 0, stream>>>(wv, wv_b, nw);
  cvt_f32_bf16<<<cg, 256, 0, stream>>>(wo, wo_b, nw);
  cvt_f32_bf16<<<cg, 256, 0, stream>>>(wf, wf_b, nw);
  int nwo = OUTd * Dm;
  cvt_f32_bf16<<<dim3((nwo + 255) / 256), 256, 0, stream>>>(w_out, wout_b, nwo);

  // h = x + pe
  int ntot = BS * Dm;
  add_pe<<<dim3((ntot + 255) / 256), 256, 0, stream>>>(x, pe, h, ntot);

  dim3 gemmGridD(Dm / 64, BS / 128);             // N=1024
  dim3 attnGrid(Sq / 16, Bsz * Hh);

  for (int l = 0; l < Ll; ++l) {
    const size_t wofs = (size_t)l * Dm * Dm;
    layernorm_bf16<<<BS, 256, 0, stream>>>(h, ln1_g + l * Dm, ln1_b + l * Dm, hn);
    gemm_wmma<<<gemmGridD, 256, 0, stream>>>(hn, wq_b + wofs, nullptr, qb, BS, Dm, Dm, 1);
    gemm_wmma<<<gemmGridD, 256, 0, stream>>>(hn, wk_b + wofs, nullptr, kb, BS, Dm, Dm, 1);
    gemm_wmma<<<gemmGridD, 256, 0, stream>>>(hn, wv_b + wofs, nullptr, vT, BS, Dm, Dm, 2);
    attn_wmma<<<attnGrid, 32, 0, stream>>>(qb, kb, vT, attnb);
    gemm_wmma<<<gemmGridD, 256, 0, stream>>>(attnb, wo_b + wofs, h, h, BS, Dm, Dm, 0);
    layernorm_bf16<<<BS, 256, 0, stream>>>(h, ln2_g + l * Dm, ln2_b + l * Dm, hn);
    gemm_wmma<<<gemmGridD, 256, 0, stream>>>(hn, wf_b + wofs, h, h, BS, Dm, Dm, 0);
  }

  layernorm_bf16<<<BS, 256, 0, stream>>>(h, lnf_g, lnf_b, hn);
  gemm_wmma<<<dim3(OUTd / 64, BS / 128), 256, 0, stream>>>(hn, wout_b, nullptr, out,
                                                           BS, OUTd, Dm, 3);
  (void)in_sizes; (void)n_in; (void)out_size; (void)ws_size;
}
